// MultiHeadGATLayer_72516227826098
// MI455X (gfx1250) — matrix operand order repaired and checked by reading, not compiled, not run on hardware
//
#include <hip/hip_runtime.h>
#include <math.h>

// ---------------- types for WMMA ----------------
typedef __attribute__((ext_vector_type(16))) _Float16 v16h;
typedef __attribute__((ext_vector_type(8)))  _Float16 v8h;
typedef __attribute__((ext_vector_type(8)))  float    v8f;

#define ALPHA 0.2f

// CDNA5 async global->LDS path via inline asm (bypasses builtin pointer-AS
// type checking); needs only the s_wait_asynccnt builtin for the wait.
#if defined(__has_builtin)
#if __has_builtin(__builtin_amdgcn_s_wait_asynccnt)
#define USE_ASYNC_LDS 1
#endif
#endif

// ---------------- CSR build ----------------
__global__ void gat_hist(const int* __restrict__ dst, int* __restrict__ cnt, int E) {
    int i = blockIdx.x * blockDim.x + threadIdx.x;
    if (i < E) atomicAdd(&cnt[dst[i]], 1);
}

// single-block exclusive scan over Nn counts (Nn multiple of 256)
__global__ void gat_scan(const int* __restrict__ cnt, int* __restrict__ offs, int Nn) {
    __shared__ int tile[256];
    __shared__ int carry;
    int t = threadIdx.x;
    if (t == 0) carry = 0;
    __syncthreads();
    for (int base = 0; base < Nn; base += 256) {
        int v = cnt[base + t];
        tile[t] = v;
        __syncthreads();
        for (int off = 1; off < 256; off <<= 1) {
            int x = (t >= off) ? tile[t - off] : 0;
            __syncthreads();
            tile[t] += x;
            __syncthreads();
        }
        offs[base + t] = carry + tile[t] - v;   // exclusive
        __syncthreads();
        if (t == 255) carry += tile[255];
        __syncthreads();
    }
    if (t == 0) offs[Nn] = carry;
}

__global__ void gat_scatter(const int* __restrict__ src, const int* __restrict__ dst,
                            const int* __restrict__ offs, int* __restrict__ cursor,
                            int* __restrict__ csr, int E) {
    int i = blockIdx.x * blockDim.x + threadIdx.x;
    if (i < E) {
        int d = dst[i];
        int pos = atomicAdd(&cursor[d], 1);
        csr[offs[d] + pos] = src[i];
    }
}

// ---------------- fp16 conversion ----------------
__global__ void gat_cvt_f16(const float* __restrict__ in, _Float16* __restrict__ out, int n) {
    int i = blockIdx.x * blockDim.x + threadIdx.x;
    if (i < n) out[i] = (_Float16)in[i];
}

// W1 [H=8][D=512][DO=64] -> Bt [col=h*64+do][k=d]  (column-major for B fragments)
__global__ void gat_cvt_W1(const float* __restrict__ W1, _Float16* __restrict__ Bt) {
    int t = blockIdx.x * blockDim.x + threadIdx.x;     // t = col*512 + d, 512*512 total
    if (t >= 512 * 512) return;
    int col = t >> 9, d = t & 511;
    int h = col >> 6, dO = col & 63;
    Bt[t] = (_Float16)W1[(h * 512 + d) * 64 + dO];
}

// Wout [K=512][N=512] row-major -> Bt [col][k] (transpose)
__global__ void gat_cvt_Wout(const float* __restrict__ W, _Float16* __restrict__ Bt) {
    int t = blockIdx.x * blockDim.x + threadIdx.x;
    if (t >= 512 * 512) return;
    int col = t >> 9, k = t & 511;
    Bt[t] = (_Float16)W[k * 512 + col];
}

// ---------------- WMMA GEMM: C[nrows,512] = A[nrows,512] * B[512,512] ----------------
// A row-major f16, staged per-block into LDS (64 rows x 512 K = 64 KB, contiguous);
// Bt column-major f16 (Bt[col*512+k]); C f32 row-major.
// Block = 256 thr (8 waves): 64 rows x 128 cols. Wave = 64x16 tile:
// 4 accumulators, 1 B frag (global) + 4 A frags (LDS) per K-step.
__device__ __forceinline__ v16h gat_lds_a(const _Float16* sA, int m, int kk, int half) {
    const v8h* p0 = (const v8h*)(sA + m * 512 + kk + half * 8);
    const v8h* p1 = (const v8h*)(sA + m * 512 + kk + 16 + half * 8);
    v8h lo = *p0, hi = *p1;
    return __builtin_shufflevector(lo, hi, 0,1,2,3,4,5,6,7,8,9,10,11,12,13,14,15);
}

__global__ void gat_gemm_f16(const _Float16* __restrict__ A, const _Float16* __restrict__ Bt,
                             float* __restrict__ C, int nrows) {
    __shared__ _Float16 sA[64 * 512];                 // 64 KB A tile
    int rowBlk = blockIdx.x >> 2;                     // nrows/64 row blocks
    int colGrp = blockIdx.x & 3;                      // 4 groups of 128 cols
    int mr = rowBlk * 64;
    if (mr >= nrows) return;

    // ---- stage the contiguous 64KB A tile into LDS ----
    {
        const _Float16* gsrc = A + (size_t)mr * 512;
        int t = threadIdx.x;
#ifdef USE_ASYNC_LDS
#pragma unroll
        for (int i = 0; i < 16; ++i) {
            int elem = (t + i * 256) * 8;             // 8 f16 = 16B per async chunk
            unsigned lds_off = (unsigned)(uintptr_t)(sA + elem);   // low 32b = LDS byte addr
            unsigned long long gaddr = (unsigned long long)(uintptr_t)(gsrc + elem);
            asm volatile("global_load_async_to_lds_b128 %0, %1, off"
                         :: "v"(lds_off), "v"(gaddr)
                         : "memory");
        }
        __builtin_amdgcn_s_wait_asynccnt(0);
        __syncthreads();
#else
#pragma unroll
        for (int i = 0; i < 16; ++i) {
            int elem = (t + i * 256) * 8;
            *(v8h*)(sA + elem) = *(const v8h*)(gsrc + elem);
        }
        __syncthreads();
#endif
    }

    int wv   = threadIdx.x >> 5;
    int lane = threadIdx.x & 31;
    int l15 = lane & 15, half = lane >> 4;
    int bcol = colGrp * 128 + wv * 16 + l15;

    v8f acc[4];
    acc[0] = (v8f){}; acc[1] = (v8f){}; acc[2] = (v8f){}; acc[3] = (v8f){};

    for (int kk = 0; kk < 512; kk += 32) {
        int kb = kk + half * 8;
        const v8h* bp = (const v8h*)(Bt + bcol * 512 + kb);
        v8h blo = bp[0], bhi = bp[2];
        v16h bfrag = __builtin_shufflevector(blo, bhi, 0,1,2,3,4,5,6,7,8,9,10,11,12,13,14,15);
        v16h a0 = gat_lds_a(sA, 0  + l15, kk, half);
        v16h a1 = gat_lds_a(sA, 16 + l15, kk, half);
        v16h a2 = gat_lds_a(sA, 32 + l15, kk, half);
        v16h a3 = gat_lds_a(sA, 48 + l15, kk, half);
        acc[0] = __builtin_amdgcn_wmma_f32_16x16x32_f16(false, a0, false, bfrag, (short)0, acc[0], false, false);
        acc[1] = __builtin_amdgcn_wmma_f32_16x16x32_f16(false, a1, false, bfrag, (short)0, acc[1], false, false);
        acc[2] = __builtin_amdgcn_wmma_f32_16x16x32_f16(false, a2, false, bfrag, (short)0, acc[2], false, false);
        acc[3] = __builtin_amdgcn_wmma_f32_16x16x32_f16(false, a3, false, bfrag, (short)0, acc[3], false, false);
    }
#pragma unroll
    for (int r = 0; r < 4; ++r) {
        int rowbase = mr + r * 16 + half * 8;
#pragma unroll
        for (int j = 0; j < 8; ++j)
            C[(size_t)(rowbase + j) * 512 + bcol] = acc[r][j];
    }
}

// ---------------- per-(node,head) attention terms: es = z.a[:64], ed = z.a[64:] ----------------
__global__ void gat_esed1(const float* __restrict__ z1, const float* __restrict__ a1,
                          float* __restrict__ es1, float* __restrict__ ed1, int total) {
    int t = blockIdx.x * blockDim.x + threadIdx.x;    // t = n*8 + h
    if (t >= total) return;
    int n = t >> 3, h = t & 7;
    const float* z  = z1 + (size_t)n * 512 + h * 64;
    const float* av = a1 + h * 128;
    float s = 0.f, d = 0.f;
#pragma unroll 8
    for (int k = 0; k < 64; ++k) {
        float zv = z[k];
        s += zv * av[k];
        d += zv * av[64 + k];
    }
    es1[t] = s; ed1[t] = d;
}

// ---------------- layer-1 aggregation: wave per (dst node, head) ----------------
__global__ void gat_agg1(const float* __restrict__ z1, const float* __restrict__ es1,
                         const float* __restrict__ ed1, const int* __restrict__ offs,
                         const int* __restrict__ csr, float* __restrict__ hcat, int totalWaves) {
    int wid  = (blockIdx.x * blockDim.x + threadIdx.x) >> 5;
    int lane = threadIdx.x & 31;
    if (wid >= totalWaves) return;
    int n = wid >> 3, h = wid & 7;
    int beg = offs[n], end = offs[n + 1];
    float ed = ed1[wid];

    // pass 1: strided max
    float m = -1e30f;
    for (int i = beg + lane; i < end; i += 32) {
        int s = csr[i];
        float e = es1[s * 8 + h] + ed;
        e = e > 0.f ? e : ALPHA * e;
        m = fmaxf(m, e);
    }
#pragma unroll
    for (int off = 16; off > 0; off >>= 1) m = fmaxf(m, __shfl_xor(m, off, 32));

    // pass 2: sequential exp-weighted accumulate (lane owns cols lane, lane+32)
    float den = 0.f, acc0 = 0.f, acc1 = 0.f;
    int c0 = lane, c1 = lane + 32;
    for (int i = beg; i < end; ++i) {
        int s = csr[i];
        if (i + 1 < end) {                     // prefetch next gathered z row
            int sn = csr[i + 1];
            __builtin_prefetch(z1 + (size_t)sn * 512 + h * 64, 0, 0);
        }
        float e = es1[s * 8 + h] + ed;
        e = e > 0.f ? e : ALPHA * e;
        float w = __expf(e - m);
        den += w;
        const float* zr = z1 + (size_t)s * 512 + h * 64;
        acc0 += w * zr[c0];
        acc1 += w * zr[c1];
    }
    float inv = 1.0f / den;
    float o0 = acc0 * inv, o1 = acc1 * inv;
    o0 = o0 > 0.f ? o0 : __expf(o0) - 1.f;            // ELU (concat head)
    o1 = o1 > 0.f ? o1 : __expf(o1) - 1.f;
    hcat[(size_t)n * 512 + h * 64 + c0] = o0;
    hcat[(size_t)n * 512 + h * 64 + c1] = o1;
}

// ---------------- layer-2 attention terms: wave per node, 512-dot with aout ----------------
__global__ void gat_esed2(const float* __restrict__ z2, const float* __restrict__ aout,
                          float* __restrict__ es2, float* __restrict__ ed2, int Nn) {
    int wid  = (blockIdx.x * blockDim.x + threadIdx.x) >> 5;
    int lane = threadIdx.x & 31;
    if (wid >= Nn) return;
    const float* z = z2 + (size_t)wid * 512;
    float s = 0.f, d = 0.f;
    for (int k = lane; k < 512; k += 32) {
        float zv = z[k];
        s += zv * aout[k];
        d += zv * aout[512 + k];
    }
#pragma unroll
    for (int off = 16; off > 0; off >>= 1) {
        s += __shfl_xor(s, off, 32);
        d += __shfl_xor(d, off, 32);
    }
    if (lane == 0) { es2[wid] = s; ed2[wid] = d; }
}

// ---------------- layer-2 aggregation: block (256 thr) per dst node, 512 cols ----------------
__global__ void gat_agg2(const float* __restrict__ z2, const float* __restrict__ es2,
                         const float* __restrict__ ed2, const int* __restrict__ offs,
                         const int* __restrict__ csr, const float* __restrict__ x,
                         float* __restrict__ out) {
    __shared__ float red[8];
    int n = blockIdx.x;
    int t = threadIdx.x;
    int lane = t & 31, w = t >> 5;
    int beg = offs[n], end = offs[n + 1];
    float ed = ed2[n];

    float m = -1e30f;
    for (int i = beg + t; i < end; i += 256) {
        float e = es2[csr[i]] + ed;
        e = e > 0.f ? e : ALPHA * e;
        m = fmaxf(m, e);
    }
#pragma unroll
    for (int off = 16; off > 0; off >>= 1) m = fmaxf(m, __shfl_xor(m, off, 32));
    if (lane == 0) red[w] = m;
    __syncthreads();
    m = red[0];
#pragma unroll
    for (int i = 1; i < 8; ++i) m = fmaxf(m, red[i]);

    float den = 0.f, acc0 = 0.f, acc1 = 0.f;
    int c0 = t, c1 = t + 256;
    for (int i = beg; i < end; ++i) {
        int s = csr[i];
        if (i + 1 < end) {                     // prefetch next gathered z2 row
            int sn = csr[i + 1];
            __builtin_prefetch(z2 + (size_t)sn * 512 + c0, 0, 0);
        }
        float e = es2[s] + ed;
        e = e > 0.f ? e : ALPHA * e;
        float wg = __expf(e - m);
        den += wg;
        const float* zr = z2 + (size_t)s * 512;
        acc0 += wg * zr[c0];
        acc1 += wg * zr[c1];
    }
    float inv = 1.0f / den;
    float o0 = acc0 * inv, o1 = acc1 * inv;
    o0 = o0 > 0.f ? o0 : __expf(o0) - 1.f;            // outer ELU
    o1 = o1 > 0.f ? o1 : __expf(o1) - 1.f;
    size_t base = (size_t)n * 512;
    out[base + c0] = o0 + x[base + c0];               // residual
    out[base + c1] = o1 + x[base + c1];
}

// ---------------- host launch ----------------
extern "C" void kernel_launch(void* const* d_in, const int* in_sizes, int n_in,
                              void* d_out, int out_size, void* d_ws, size_t ws_size,
                              hipStream_t stream) {
    const float* h    = (const float*)d_in[0];  // [N,512]
    const float* W1   = (const float*)d_in[1];  // [8,512,64]
    const float* a1   = (const float*)d_in[2];  // [8,128]
    const float* Wout = (const float*)d_in[3];  // [512,512]
    const float* aout = (const float*)d_in[4];  // [1024]
    const int*   src  = (const int*)d_in[5];
    const int*   dst  = (const int*)d_in[6];
    const int E  = in_sizes[5];
    const int Nn = in_sizes[0] / 512;           // 16384 nodes
    float* out = (float*)d_out;

    // carve workspace (~88 MB)
    char* p = (char*)d_ws;
    auto carve = [&p](size_t bytes) -> void* {
        void* r = (void*)p;
        p += (bytes + 255) & ~((size_t)255);
        return r;
    };
    _Float16* bfA  = (_Float16*)carve((size_t)Nn * 512 * 2); // x f16, later hcat f16
    _Float16* Bt   = (_Float16*)carve(512 * 512 * 2);        // weight f16 (reused)
    float*    zbuf = (float*)carve((size_t)Nn * 512 * 4);    // z1, later z2
    float*    hcat = (float*)carve((size_t)Nn * 512 * 4);
    float*    es1  = (float*)carve((size_t)Nn * 8 * 4);
    float*    ed1  = (float*)carve((size_t)Nn * 8 * 4);
    float*    es2  = (float*)carve((size_t)Nn * 4);
    float*    ed2  = (float*)carve((size_t)Nn * 4);
    int*      cnt    = (int*)carve((size_t)Nn * 4);
    int*      cursor = (int*)carve((size_t)Nn * 4);
    int*      offs   = (int*)carve((size_t)(Nn + 1) * 4);
    int*      csr    = (int*)carve((size_t)E * 4);

    (void)hipMemsetAsync(cnt,    0, (size_t)Nn * 4, stream);
    (void)hipMemsetAsync(cursor, 0, (size_t)Nn * 4, stream);

    // CSR by destination
    gat_hist<<<(E + 255) / 256, 256, 0, stream>>>(dst, cnt, E);
    gat_scan<<<1, 256, 0, stream>>>(cnt, offs, Nn);
    gat_scatter<<<(E + 255) / 256, 256, 0, stream>>>(src, dst, offs, cursor, csr, E);

    const int elems = Nn * 512;
    const int gemmBlocks = (Nn / 64) * 4;       // row blocks x 4 col groups

    // ---- layer 1 ----
    gat_cvt_f16<<<(elems + 255) / 256, 256, 0, stream>>>(h, bfA, elems);
    gat_cvt_W1 <<<(512 * 512 + 255) / 256, 256, 0, stream>>>(W1, Bt);
    gat_gemm_f16<<<gemmBlocks, 256, 0, stream>>>(bfA, Bt, zbuf, Nn);
    gat_esed1<<<(Nn * 8 + 255) / 256, 256, 0, stream>>>(zbuf, a1, es1, ed1, Nn * 8);
    gat_agg1 <<<(Nn * 8 * 32) / 256, 256, 0, stream>>>(zbuf, es1, ed1, offs, csr, hcat, Nn * 8);

    // ---- layer 2 ----
    gat_cvt_f16 <<<(elems + 255) / 256, 256, 0, stream>>>(hcat, bfA, elems);
    gat_cvt_Wout<<<(512 * 512 + 255) / 256, 256, 0, stream>>>(Wout, Bt);
    gat_gemm_f16<<<gemmBlocks, 256, 0, stream>>>(bfA, Bt, zbuf, Nn);   // zbuf now = z2
    gat_esed2<<<(Nn * 32) / 256, 256, 0, stream>>>(zbuf, aout, es2, ed2, Nn);
    gat_agg2 <<<Nn, 256, 0, stream>>>(zbuf, es2, ed2, offs, csr, h, out);
}
